// LocalTemporalMGMQTorchModel_5497558139092
// MI455X (gfx1250) — matrix operand order repaired and checked by reading, not compiled, not run on hardware
//
#include <hip/hip_runtime.h>
#include <hip/hip_bf16.h>

// ---------------------------------------------------------------------------
// LocalTemporalMGMQ forward for gfx1250 (MI455X, wave32, WMMA).
//
// Stage 0 (prep_weights): pre-swizzle coop_W/conf_W/sage_W into f16 WMMA
//   B-fragment layout (lane-major, 16 halves contiguous per lane) so the hot
//   kernels fetch fragments with two global_load_b128 per lane, no converts.
// Stage 1 (gat_kernel):   102400 independent 12-lane graphs. proj+LN+ReLU on
//   VALU, then 8 head-GEMMs [16(pad12),64]x[64,32] per graph via
//   v_wmma_f32_16x16x32_f16 (2 K-chunks x 2 N-tiles x 8 heads = 32 WMMA/graph),
//   masked softmax attention + ELU + lane-mean on VALU. Emb stored f16 in WS
//   (~52 MB -> L2 resident on the 192 MB L2).
// Stage 2 (sage_kernel):  node-0-only GraphSAGE: [20480,512]x[512,64] GEMM,
//   16 rows per wave, 16 K-chunks x 4 N-tiles = 64 WMMA per wave.
// Stage 3 (gru_kernel):   bidirectional GRU, one wave per batch row (lane =
//   hidden unit), recurrence on VALU (too small/serial for matrix ops).
// Stage 4 (head_kernel):  LN-MLP policy/value heads, one wave per row with
//   shuffle-reduction LayerNorm.
// ---------------------------------------------------------------------------

typedef _Float16 v16h __attribute__((ext_vector_type(16)));
typedef _Float16 v8h  __attribute__((ext_vector_type(8)));
typedef float    v8f  __attribute__((ext_vector_type(8)));

#define NB    4096      // batch
#define NT    5         // time steps
#define NK    4         // neighbors
#define GSELF (NB*NT)           // 20480 self graphs
#define GNB   (NB*NK*NT)        // 81920 neighbor graphs
#define GTOT  (GSELF+GNB)       // 102400

#define GATF_ELEMS  (16*32*16)   // 16 B-frags (2 gat x 4 head x 2 kc x 2 nt)
#define SAGEF_ELEMS (64*32*16)   // 64 B-frags (16 kc x 4 nt)

// --------------------- Stage 0: weight pre-swizzle (f16) -------------------
// B-fragment layout per ISA 7.12.2: lanes 0-15 -> N=lane, K 0-7,16-23;
// lanes 16-31 -> N=lane-16, K 8-15,24-31 (within a 32-K chunk). Fragment stored
// lane-major: frag[f][lane][j], j<8 -> K 8h+j, j>=8 -> K 16+8h+(j-8).
__global__ __launch_bounds__(256) void prep_weights(
    const float* __restrict__ coopW, const float* __restrict__ confW,
    const float* __restrict__ sW,
    _Float16* __restrict__ gatWf, _Float16* __restrict__ sageWf)
{
    int tid = blockIdx.x * 256 + threadIdx.x;
    if (tid < GATF_ELEMS) {
        int j  = tid & 15;
        int L  = (tid >> 4) & 31;
        int f  = tid >> 9;
        int nt = f & 1, kc = (f >> 1) & 1, hh = (f >> 2) & 3, gt = (f >> 4) & 1;
        const float* W = (gt ? confW : coopW) + hh * 64 * 32;
        int kr  = kc*32 + ((L >= 16) ? 8 : 0) + (j < 8 ? j : j + 8);
        int col = nt*16 + (L & 15);
        gatWf[tid] = (_Float16)W[kr*32 + col];
    } else {
        int sid = tid - GATF_ELEMS;
        if (sid < SAGEF_ELEMS) {
            int j  = sid & 15;
            int L  = (sid >> 4) & 31;
            int f  = sid >> 9;
            int nt = f & 3, kc = f >> 2;
            int kr  = kc*32 + ((L >= 16) ? 8 : 0) + (j < 8 ? j : j + 8);
            int col = nt*16 + (L & 15);
            sageWf[sid] = (_Float16)sW[kr*64 + col];
        }
    }
}

// ------------------------------ Stage 1: GAT -------------------------------
// 4 waves / block, one wave per graph.
__global__ __launch_bounds__(128) void gat_kernel(
    const float* __restrict__ selfF, const float* __restrict__ nbF,
    const float* __restrict__ pW,  const float* __restrict__ pb,
    const float* __restrict__ pg,  const float* __restrict__ pbt,
    const float* __restrict__ coopSrc, const float* __restrict__ coopDst,
    const float* __restrict__ confSrc, const float* __restrict__ confDst,
    const _Float16* __restrict__ gatWf,
    _Float16* __restrict__ embS, _Float16* __restrict__ embN)
{
    const int wave = threadIdx.x >> 5;
    const int lane = threadIdx.x & 31;
    const int g = blockIdx.x * 4 + wave;

    const float* xrow;
    _Float16*    dst;
    if (g < GSELF) { xrow = selfF + (size_t)g * 48;          dst = embS + (size_t)g * 256; }
    else           { int g2 = g - GSELF;
                     xrow = nbF + (size_t)g2 * 48;           dst = embN + (size_t)g2 * 256; }
    __builtin_prefetch(xrow, 0, 3);   // global_prefetch_b8

    __shared__ alignas(16) _Float16 sh_h [4][12 * 64];  // projected node feats
    __shared__ float    sh_Wh[4][16 * 32];   // per-head Wh staging
    __shared__ float    sh_at[4][12 * 12];   // attention weights
    __shared__ float    sh_s [4][32];        // [0..15]=src score, [16..31]=dst

    _Float16* hT   = sh_h [wave];
    float*    WhS  = sh_Wh[wave];
    float*    attS = sh_at[wave];
    float*    sS   = sh_s [wave];

    // ---- projection [12,4]@[4,64] + LayerNorm + ReLU, 2 channels per lane ----
    for (int n = 0; n < 12; ++n) {
        float x0 = xrow[n*4+0], x1 = xrow[n*4+1], x2 = xrow[n*4+2], x3 = xrow[n*4+3];
        int c1 = lane, c2 = lane + 32;
        float a1 = pb[c1] + x0*pW[c1] + x1*pW[64+c1] + x2*pW[128+c1] + x3*pW[192+c1];
        float a2 = pb[c2] + x0*pW[c2] + x1*pW[64+c2] + x2*pW[128+c2] + x3*pW[192+c2];
        float s = a1 + a2, q = a1*a1 + a2*a2;
        for (int m = 16; m >= 1; m >>= 1) { s += __shfl_xor(s, m, 32); q += __shfl_xor(q, m, 32); }
        float mean = s * (1.0f/64.0f);
        float var  = q * (1.0f/64.0f) - mean*mean;
        float rs   = rsqrtf(var + 1e-5f);
        float h1 = (a1 - mean)*rs*pg[c1] + pbt[c1]; h1 = h1 > 0.f ? h1 : 0.f;
        float h2 = (a2 - mean)*rs*pg[c2] + pbt[c2]; h2 = h2 > 0.f ? h2 : 0.f;
        hT[n*64 + c1] = (_Float16)h1;
        hT[n*64 + c2] = (_Float16)h2;
    }
    __syncthreads();

    // ---- A fragments (16x32 f16 layout, ISA 7.12.2), two ds_load_b128 each ----
    const int am  = lane & 15;
    const int aok = (lane >= 16) ? 8 : 0;
    v16h afrag[2];
    for (int kc = 0; kc < 2; ++kc) {
        int ko = kc*32 + aok;
        if (am < 12) {
            v8h lo = *(const v8h*)(hT + am*64 + ko);
            v8h hi = *(const v8h*)(hT + am*64 + ko + 16);
            for (int j = 0; j < 8; ++j) { afrag[kc][j] = lo[j]; afrag[kc][j+8] = hi[j]; }
        } else {
            for (int j = 0; j < 16; ++j) afrag[kc][j] = (_Float16)0.0f;
        }
    }

    const v16h* gatWfV = (const v16h*)gatWf;
    const float inv12 = 1.0f / 12.0f;

    for (int gt = 0; gt < 2; ++gt) {            // 0 = coop GAT, 1 = conf GAT
        const float* Asr = gt ? confSrc : coopSrc;
        const float* Ads = gt ? confDst : coopDst;
        for (int hh = 0; hh < 4; ++hh) {
            v8f acc0 = {}, acc1 = {};
            for (int kc = 0; kc < 2; ++kc) {
                int fb = ((gt*4 + hh)*2 + kc)*2;     // fragment base index
                v16h b0 = gatWfV[(fb + 0)*32 + lane];   // 2x global_load_b128
                v16h b1 = gatWfV[(fb + 1)*32 + lane];
                acc0 = __builtin_amdgcn_wmma_f32_16x16x32_f16(false, afrag[kc], false, b0, (short)0, acc0, false, false);
                acc1 = __builtin_amdgcn_wmma_f32_16x16x32_f16(false, afrag[kc], false, b1, (short)0, acc1, false, false);
            }
            // stage Wh tile (C layout: lanes0-15 -> M=r, lanes16-31 -> M=r+8)
            for (int r = 0; r < 8; ++r) {
                int row = (lane < 16) ? r : r + 8;
                WhS[row*32 + (lane & 15)]      = acc0[r];
                WhS[row*32 + 16 + (lane & 15)] = acc1[r];
            }
            __syncthreads();

            // attention scores: s_src[n], s_dst[n]
            if (lane < 12) {
                float ss = 0.f, sd = 0.f;
                for (int o = 0; o < 32; ++o) {
                    float w = WhS[lane*32 + o];
                    ss += w * Asr[hh*32 + o];
                    sd += w * Ads[hh*32 + o];
                }
                sS[lane] = ss; sS[16 + lane] = sd;
            }
            __syncthreads();

            // masked leaky-relu + softmax (hardcoded lane adjacency)
            if (lane < 12) {
                int n = lane, blk = n / 3;
                float e[12], mx = -1e30f;
                for (int m = 0; m < 12; ++m) {
                    bool adj = gt ? ((m/3 != blk) || (m == n)) : (m/3 == blk);
                    if (adj) {
                        float v = sS[n] + sS[16 + m];
                        v = v > 0.f ? v : 0.2f * v;
                        e[m] = v; if (v > mx) mx = v;
                    } else e[m] = -1e30f;
                }
                float sum = 0.f;
                for (int m = 0; m < 12; ++m) {
                    float v = (e[m] > -1e29f) ? __expf(e[m] - mx) : 0.f;
                    e[m] = v; sum += v;
                }
                float inv = 1.0f / sum;
                for (int m = 0; m < 12; ++m) attS[n*12 + m] = e[m] * inv;
            }
            __syncthreads();

            // out = att @ Wh, ELU, mean over 12 nodes; lane = output channel
            {
                int o = lane;
                float pool = 0.f;
                for (int n = 0; n < 12; ++n) {
                    float v = 0.f;
                    for (int m = 0; m < 12; ++m) v += attS[n*12 + m] * WhS[m*32 + o];
                    v = v > 0.f ? v : (__expf(v) - 1.0f);
                    pool += v;
                }
                dst[gt*128 + hh*32 + o] = (_Float16)(pool * inv12);
            }
            __syncthreads();
        }
    }
}

// --------------------------- Stage 2: GraphSAGE ----------------------------
// Only node 0 is consumed downstream. 2 waves / block, 16 rows per wave.
__global__ __launch_bounds__(64) void sage_kernel(
    const _Float16* __restrict__ embS, const _Float16* __restrict__ embN,
    const float* __restrict__ mask, const _Float16* __restrict__ sageWf,
    const float* __restrict__ sb, float* __restrict__ hsage)
{
    const int wave = threadIdx.x >> 5;
    const int lane = threadIdx.x & 31;
    const int r0 = (blockIdx.x * 2 + wave) * 16;

    __shared__ alignas(16) _Float16 sh_x[2][16 * 512];
    _Float16* X = sh_x[wave];

    for (int j = 0; j < 16; ++j) {
        int row = r0 + j;
        int b = row / NT, t = row % NT;
        float m0 = mask[b*4+0], m1 = mask[b*4+1], m2 = mask[b*4+2], m3 = mask[b*4+3];
        float deg = m0 + m1 + m2 + m3; deg = deg > 1.0f ? deg : 1.0f;
        float invd = 1.0f / deg;
        for (int c = lane; c < 512; c += 32) {
            float v;
            if (c < 256) {
                v = (float)embS[(size_t)row*256 + c];
            } else {
                int cc = c - 256;
                size_t base = ((size_t)b*NK*NT + t) * 256 + cc;
                v = ( m0*(float)embN[base]
                    + m1*(float)embN[base + 1*NT*256]
                    + m2*(float)embN[base + 2*NT*256]
                    + m3*(float)embN[base + 3*NT*256]) * invd;
            }
            X[j*512 + c] = (_Float16)v;
        }
    }
    __syncthreads();

    const int am  = lane & 15;
    const int aok = (lane >= 16) ? 8 : 0;
    const v16h* sageWfV = (const v16h*)sageWf;

    v8f acc[4] = {};
    for (int kc = 0; kc < 16; ++kc) {
        int ko = kc*32 + aok;
        v16h a;
        {
            v8h lo = *(const v8h*)(X + am*512 + ko);
            v8h hi = *(const v8h*)(X + am*512 + ko + 16);
            for (int j = 0; j < 8; ++j) { a[j] = lo[j]; a[j+8] = hi[j]; }
        }
        for (int nt = 0; nt < 4; ++nt) {
            v16h bf = sageWfV[(kc*4 + nt)*32 + lane];
            acc[nt] = __builtin_amdgcn_wmma_f32_16x16x32_f16(false, a, false, bf, (short)0, acc[nt], false, false);
        }
    }
    for (int nt = 0; nt < 4; ++nt) {
        for (int r = 0; r < 8; ++r) {
            int row = r0 + ((lane < 16) ? r : r + 8);
            int o = nt*16 + (lane & 15);
            float v = acc[nt][r] + sb[o];
            hsage[(size_t)row*64 + o] = v > 0.f ? v : 0.f;
        }
    }
}

// ------------------------- Stage 3: bidirectional GRU ----------------------
// One wave per batch element, lane = hidden unit (GRU_H = 32 = wave32).
__global__ __launch_bounds__(256) void gru_kernel(
    const float* __restrict__ hsage,
    const float* __restrict__ fWx, const float* __restrict__ fWh,
    const float* __restrict__ fbx, const float* __restrict__ fbh,
    const float* __restrict__ bWx, const float* __restrict__ bWh,
    const float* __restrict__ bbx, const float* __restrict__ bbh,
    float* __restrict__ net)
{
    const int wave = threadIdx.x >> 5;
    const int lane = threadIdx.x & 31;
    const int b = blockIdx.x * 8 + wave;

    __shared__ float sh_x[8][64];
    __shared__ float sh_h[8][32];
    float* X = sh_x[wave];
    float* H = sh_h[wave];

    for (int dir = 0; dir < 2; ++dir) {
        const float* Wx = dir ? bWx : fWx;
        const float* Wh = dir ? bWh : fWh;
        const float* bx = dir ? bbx : fbx;
        const float* bh = dir ? bbh : fbh;
        H[lane] = 0.0f;
        __syncthreads();
        for (int s = 0; s < NT; ++s) {
            int t = dir ? (NT - 1 - s) : s;
            X[lane]      = hsage[((size_t)b*NT + t)*64 + lane];
            X[lane + 32] = hsage[((size_t)b*NT + t)*64 + lane + 32];
            __syncthreads();
            int j = lane;
            float gxr = bx[j], gxz = bx[32+j], gxn = bx[64+j];
            for (int i = 0; i < 64; ++i) {
                float xi = X[i];
                gxr += xi * Wx[i*96 + j];
                gxz += xi * Wx[i*96 + 32 + j];
                gxn += xi * Wx[i*96 + 64 + j];
            }
            float ghr = bh[j], ghz = bh[32+j], ghn = bh[64+j];
            for (int i = 0; i < 32; ++i) {
                float hi = H[i];
                ghr += hi * Wh[i*96 + j];
                ghz += hi * Wh[i*96 + 32 + j];
                ghn += hi * Wh[i*96 + 64 + j];
            }
            float rg = 1.0f / (1.0f + __expf(-(gxr + ghr)));
            float zg = 1.0f / (1.0f + __expf(-(gxz + ghz)));
            float ng = tanhf(gxn + rg * ghn);
            float hn = (1.0f - zg)*ng + zg*H[j];
            __syncthreads();
            H[j] = hn;
            __syncthreads();
        }
        net[(size_t)b*64 + dir*32 + lane] = H[lane];
        __syncthreads();
    }
}

// ---------------------- Stage 4: LN-MLP policy/value heads -----------------
// One wave per batch row; LayerNorm via shuffle reductions.
__global__ __launch_bounds__(256) void head_kernel(
    const _Float16* __restrict__ embS, const float* __restrict__ net,
    const float* __restrict__ pW1, const float* __restrict__ pb1, const float* __restrict__ pg1, const float* __restrict__ pbt1,
    const float* __restrict__ pW2, const float* __restrict__ pb2, const float* __restrict__ pg2, const float* __restrict__ pbt2,
    const float* __restrict__ pWo, const float* __restrict__ pbo,
    const float* __restrict__ vW1, const float* __restrict__ vb1, const float* __restrict__ vg1, const float* __restrict__ vbt1,
    const float* __restrict__ vW2, const float* __restrict__ vb2, const float* __restrict__ vg2, const float* __restrict__ vbt2,
    const float* __restrict__ vWo, const float* __restrict__ vbo,
    float* __restrict__ out)
{
    const int wave = threadIdx.x >> 5;
    const int lane = threadIdx.x & 31;
    const int b = blockIdx.x * 8 + wave;

    __shared__ float sh_j [8][320];
    __shared__ float sh_y1[8][128];
    __shared__ float sh_y2[8][64];
    float* J  = sh_j [wave];
    float* Y1 = sh_y1[wave];
    float* Y2 = sh_y2[wave];

    // joint = concat(self_emb[:, T-1, :], net)
    for (int c = lane; c < 320; c += 32)
        J[c] = (c < 256) ? (float)embS[((size_t)b*NT + (NT-1))*256 + c]
                         : net[(size_t)b*64 + (c - 256)];
    __syncthreads();

    for (int head = 0; head < 2; ++head) {
        const float* W1  = head ? vW1  : pW1;  const float* b1  = head ? vb1  : pb1;
        const float* g1  = head ? vg1  : pg1;  const float* bt1 = head ? vbt1 : pbt1;
        const float* W2  = head ? vW2  : pW2;  const float* b2  = head ? vb2  : pb2;
        const float* g2  = head ? vg2  : pg2;  const float* bt2 = head ? vbt2 : pbt2;
        const float* Wo  = head ? vWo  : pWo;  const float* bo  = head ? vbo  : pbo;

        // layer 1: 320 -> 128 (+LN+ReLU), 4 outputs per lane
        float o1[4]; float s = 0.f, q = 0.f;
        for (int u = 0; u < 4; ++u) {
            int o = lane*4 + u;
            float acc = b1[o];
            for (int i = 0; i < 320; ++i) acc += J[i] * W1[i*128 + o];
            o1[u] = acc; s += acc; q += acc*acc;
        }
        for (int m = 16; m >= 1; m >>= 1) { s += __shfl_xor(s, m, 32); q += __shfl_xor(q, m, 32); }
        float mean = s * (1.0f/128.0f), var = q * (1.0f/128.0f) - mean*mean;
        float rs = rsqrtf(var + 1e-5f);
        for (int u = 0; u < 4; ++u) {
            int o = lane*4 + u;
            float v = (o1[u] - mean)*rs*g1[o] + bt1[o];
            Y1[o] = v > 0.f ? v : 0.f;
        }
        __syncthreads();

        // layer 2: 128 -> 64 (+LN+ReLU), 2 outputs per lane
        float o2[2]; s = 0.f; q = 0.f;
        for (int u = 0; u < 2; ++u) {
            int o = lane*2 + u;
            float acc = b2[o];
            for (int i = 0; i < 128; ++i) acc += Y1[i] * W2[i*64 + o];
            o2[u] = acc; s += acc; q += acc*acc;
        }
        for (int m = 16; m >= 1; m >>= 1) { s += __shfl_xor(s, m, 32); q += __shfl_xor(q, m, 32); }
        mean = s * (1.0f/64.0f); var = q * (1.0f/64.0f) - mean*mean;
        rs = rsqrtf(var + 1e-5f);
        for (int u = 0; u < 2; ++u) {
            int o = lane*2 + u;
            float v = (o2[u] - mean)*rs*g2[o] + bt2[o];
            Y2[o] = v > 0.f ? v : 0.f;
        }
        __syncthreads();

        // output layer
        int nout = head ? 1 : 8;
        if (lane < nout) {
            float acc = bo[lane];
            for (int i = 0; i < 64; ++i) acc += Y2[i] * Wo[i*nout + lane];
            if (head) out[(size_t)NB*8 + b] = acc;     // value
            else      out[(size_t)b*8 + lane] = acc;   // logits
        }
        __syncthreads();
    }
}

// ------------------------------- launcher ----------------------------------
extern "C" void kernel_launch(void* const* d_in, const int* in_sizes, int n_in,
                              void* d_out, int out_size, void* d_ws, size_t ws_size,
                              hipStream_t stream) {
    const float* selfF   = (const float*)d_in[0];
    const float* nbF     = (const float*)d_in[1];
    const float* nmask   = (const float*)d_in[2];
    const float* pW      = (const float*)d_in[3];
    const float* pb      = (const float*)d_in[4];
    const float* pg      = (const float*)d_in[5];
    const float* pbt     = (const float*)d_in[6];
    const float* coopW   = (const float*)d_in[7];
    const float* coopSrc = (const float*)d_in[8];
    const float* coopDst = (const float*)d_in[9];
    const float* confW   = (const float*)d_in[10];
    const float* confSrc = (const float*)d_in[11];
    const float* confDst = (const float*)d_in[12];
    const float* sW      = (const float*)d_in[13];
    const float* sb      = (const float*)d_in[14];
    const float* fWx     = (const float*)d_in[15];
    const float* fWh     = (const float*)d_in[16];
    const float* fbx     = (const float*)d_in[17];
    const float* fbh     = (const float*)d_in[18];
    const float* bWx     = (const float*)d_in[19];
    const float* bWh     = (const float*)d_in[20];
    const float* bbx     = (const float*)d_in[21];
    const float* bbh     = (const float*)d_in[22];
    const float* pW1     = (const float*)d_in[23];
    const float* pb1     = (const float*)d_in[24];
    const float* pg1     = (const float*)d_in[25];
    const float* pbt1    = (const float*)d_in[26];
    const float* pW2     = (const float*)d_in[27];
    const float* pb2     = (const float*)d_in[28];
    const float* pg2     = (const float*)d_in[29];
    const float* pbt2    = (const float*)d_in[30];
    const float* pWo     = (const float*)d_in[31];
    const float* pbo     = (const float*)d_in[32];
    const float* vW1     = (const float*)d_in[33];
    const float* vb1     = (const float*)d_in[34];
    const float* vg1     = (const float*)d_in[35];
    const float* vbt1    = (const float*)d_in[36];
    const float* vW2     = (const float*)d_in[37];
    const float* vb2     = (const float*)d_in[38];
    const float* vg2     = (const float*)d_in[39];
    const float* vbt2    = (const float*)d_in[40];
    const float* vWo     = (const float*)d_in[41];
    const float* vbo     = (const float*)d_in[42];
    float* out = (float*)d_out;

    // workspace layout (halves first; all offsets 32B-aligned)
    _Float16* gatWf  = (_Float16*)d_ws;                       // 16 KB
    _Float16* sageWf = gatWf + GATF_ELEMS;                    // 64 KB
    _Float16* embS   = sageWf + SAGEF_ELEMS;                  // 10.5 MB
    _Float16* embN   = embS + (size_t)GSELF * 256;            // 42 MB
    float*    hsage  = (float*)(embN + (size_t)GNB * 256);    // 5.2 MB
    float*    net    = hsage + (size_t)NB * NT * 64;          // 1 MB

    prep_weights<<<(GATF_ELEMS + SAGEF_ELEMS) / 256, 256, 0, stream>>>(
        coopW, confW, sW, gatWf, sageWf);

    gat_kernel<<<GTOT / 4, 128, 0, stream>>>(
        selfF, nbF, pW, pb, pg, pbt,
        coopSrc, coopDst, confSrc, confDst, gatWf, embS, embN);

    sage_kernel<<<(NB * NT) / 32, 64, 0, stream>>>(embS, embN, nmask, sageWf, sb, hsage);

    gru_kernel<<<NB / 8, 256, 0, stream>>>(hsage, fWx, fWh, fbx, fbh,
                                           bWx, bWh, bbx, bbh, net);

    head_kernel<<<NB / 8, 256, 0, stream>>>(embS, net,
        pW1, pb1, pg1, pbt1, pW2, pb2, pg2, pbt2, pWo, pbo,
        vW1, vb1, vg1, vbt1, vW2, vb2, vg2, vbt2, vWo, vbo, out);
}
